// GAT_Close_79783312490615
// MI455X (gfx1250) — compile-verified, matrix operand order checked by reading, not run on hardware
//
#include <hip/hip_runtime.h>
#include <hip/hip_bf16.h>

#define NN      100000          /* divisible by 16: row-tile grids are exact */
#define NHID    128
#define NE1     1600000
#define NE2     1600000
#define ETOT    (NE2 + NN)      /* edges + self loops */
#define NGAT    7
#define NINTER  6
#define NEG_SLOPE 0.2f
#define EPSN    1e-12f

typedef __attribute__((ext_vector_type(16))) __bf16 v16bf;
typedef __attribute__((ext_vector_type(8)))  float  v8f;

// ---------------------------------------------------------------------------
// WMMA fragment layout (gfx1250 wave32, V_WMMA_F32_16X16X32_BF16).
// A operand 16x32 (MxK): lane holds row m = lane&15; element j holds
//   k = kk*32 + ((j>>3)<<4) + ((lane>>4)<<3) + (j&7).
// B operand 32x16 (KxN) mirrors A with column n = n0 + (lane&15).
// Fragments stored contiguously: frag[(tile*32 + lane)*16 + j], so each lane
// reads its 16 bf16 elements as one aligned 32-byte vector load (2x b128).
// ---------------------------------------------------------------------------
__device__ __forceinline__ v8f wmma_bf16(v16bf a, v16bf b, v8f c) {
  return __builtin_amdgcn_wmma_f32_16x16x32_bf16(false, a, false, b, (short)0, c,
                                                 false, false);
}

__device__ __forceinline__ v16bf frag_load(const __bf16* p) {
  return *(const v16bf*)p;
}

// order-preserving float <-> uint encoding for atomicMax-based segment_max
__device__ __forceinline__ unsigned enc_f32(float f) {
  unsigned u = __float_as_uint(f);
  return (u & 0x80000000u) ? ~u : (u | 0x80000000u);
}
__device__ __forceinline__ float dec_f32(unsigned u) {
  return (u & 0x80000000u) ? __uint_as_float(u & 0x7fffffffu)
                           : __uint_as_float(~u);
}

__device__ __forceinline__ void edge_sd(const int* __restrict__ ei2, int e,
                                        int& s, int& d) {
  if (e < NE2) { s = ei2[e]; d = ei2[NE2 + e]; }
  else         { s = e - NE2; d = s; }
}
__device__ __forceinline__ float leaky(float v) {
  return v > 0.f ? v : NEG_SLOPE * v;
}

// ---------------------------------------------------------------------------
// Weight pre-pack: W (K x C, f32 row-major) -> bf16 B-fragments.
// dst[ ((ct*(K/32)+kk)*32 + lane)*16 + j ] = bf16(W[k*C + ct*16 + (lane&15)])
// ---------------------------------------------------------------------------
__global__ void k_pack_w(const float* __restrict__ W, __bf16* __restrict__ Wf,
                         int K, int C) {
  int idx = blockIdx.x * 256 + threadIdx.x;
  if (idx >= K * C) return;
  int j    = idx & 15;
  int lane = (idx >> 4) & 31;
  int t    = idx >> 9;              // ct*(K/32) + kk
  int ksteps = K >> 5;
  int ct = t / ksteps, kk = t - ct * ksteps;
  int n = ct * 16 + (lane & 15);
  int k = kk * 32 + ((j >> 3) << 4) + ((lane >> 4) << 3) + (j & 7);
  Wf[idx] = (__bf16)W[k * C + n];
}

// ---------------------------------------------------------------------------
// Fill kernels
// ---------------------------------------------------------------------------
__global__ void k_fill_f32(float* __restrict__ p, float v, int n) {
  int i = blockIdx.x * 256 + threadIdx.x;
  if (i < n) p[i] = v;
}
__global__ void k_fill_u32(unsigned* __restrict__ p, unsigned v, int n) {
  int i = blockIdx.x * 256 + threadIdx.x;
  if (i < n) p[i] = v;
}

// ---------------------------------------------------------------------------
// x = segment_sum(weight_init[src1] * vals1, dst1)  (wave per edge, 4 f/lane)
// ---------------------------------------------------------------------------
__global__ void k_spmm(const int* __restrict__ ei1, const float* __restrict__ vals,
                       const float* __restrict__ win, float* __restrict__ x) {
  int gid  = blockIdx.x * 256 + threadIdx.x;
  int e    = gid >> 5;
  int lane = gid & 31;
  if (e >= NE1) return;
  int s = ei1[e], d = ei1[NE1 + e];
  float val = vals[e];
  const float4 w = ((const float4*)(win + (size_t)s * NHID))[lane];
  float* out = x + (size_t)d * NHID + lane * 4;
  atomicAdd(out + 0, w.x * val);
  atomicAdd(out + 1, w.y * val);
  atomicAdd(out + 2, w.z * val);
  atomicAdd(out + 3, w.w * val);
}

// ---------------------------------------------------------------------------
// H = X @ W  (X: NN x 128 f32; W pre-packed bf16 fragments, C = 128).
// Block = 256 threads (8 waves) covers 16 rows exactly (NN % 16 == 0);
// wave w covers columns 16w. A stripe staged in LDS in fragment order (4 KB).
// ---------------------------------------------------------------------------
__global__ void k_gemm128(const float* __restrict__ X, const __bf16* __restrict__ Wf,
                          float* __restrict__ H) {
  __shared__ __attribute__((aligned(32))) __bf16 Af[4 * 32 * 16];  // 4 KB
  int m0 = blockIdx.x * 16;
  int tid = threadIdx.x;
  for (int i = tid; i < 4 * 32 * 16; i += 256) {
    int j = i & 15, ln = (i >> 4) & 31, kk = i >> 9;
    int m = ln & 15;
    int k = kk * 32 + ((j >> 3) << 4) + ((ln >> 4) << 3) + (j & 7);
    Af[i] = (__bf16)X[(size_t)(m0 + m) * NHID + k];
  }
  __syncthreads();
  int wave = tid >> 5, lane = tid & 31;
  v8f acc = {};
#pragma unroll
  for (int kk = 0; kk < 4; ++kk) {
    v16bf a = frag_load(&Af[(kk * 32 + lane) * 16]);
    v16bf b = frag_load(&Wf[((size_t)(wave * 4 + kk) * 32 + lane) * 16]);
    acc = wmma_bf16(a, b, acc);
  }
  int col   = wave * 16 + (lane & 15);
  int rbase = m0 + ((lane >> 4) << 3);
  float* out = H + (size_t)rbase * NHID + col;
#pragma unroll
  for (int r = 0; r < 8; ++r)
    out[(size_t)r * NHID] = acc[r];
}

// ---------------------------------------------------------------------------
// als = H @ a_s ; ald = H @ a_d   (wave per node)
// ---------------------------------------------------------------------------
__global__ void k_att_logits(const float* __restrict__ H,
                             const float* __restrict__ a_s,
                             const float* __restrict__ a_d,
                             float* __restrict__ als, float* __restrict__ ald) {
  int gid  = blockIdx.x * 256 + threadIdx.x;
  int node = gid >> 5, lane = gid & 31;
  if (node >= NN) return;
  const float4 h   = ((const float4*)(H + (size_t)node * NHID))[lane];
  const float4 as4 = ((const float4*)a_s)[lane];
  const float4 ad4 = ((const float4*)a_d)[lane];
  float ps = h.x * as4.x + h.y * as4.y + h.z * as4.z + h.w * as4.w;
  float pd = h.x * ad4.x + h.y * ad4.y + h.z * ad4.z + h.w * ad4.w;
#pragma unroll
  for (int m = 16; m > 0; m >>= 1) {
    ps += __shfl_xor(ps, m, 32);
    pd += __shfl_xor(pd, m, 32);
  }
  if (lane == 0) { als[node] = ps; ald[node] = pd; }
}

// ---------------------------------------------------------------------------
// Edge softmax passes
// ---------------------------------------------------------------------------
__global__ void k_edge_max(const int* __restrict__ ei2, const float* __restrict__ als,
                           const float* __restrict__ ald, unsigned* __restrict__ mmax) {
  int e = blockIdx.x * 256 + threadIdx.x;
  if (e >= ETOT) return;
  int s, d; edge_sd(ei2, e, s, d);
  float v = leaky(als[s] + ald[d]);
  atomicMax(&mmax[d], enc_f32(v));
}

__global__ void k_edge_sum(const int* __restrict__ ei2, const float* __restrict__ als,
                           const float* __restrict__ ald,
                           const unsigned* __restrict__ mmax,
                           float* __restrict__ denom) {
  int e = blockIdx.x * 256 + threadIdx.x;
  if (e >= ETOT) return;
  int s, d; edge_sd(ei2, e, s, d);
  float v = leaky(als[s] + ald[d]);
  float w = __expf(v - dec_f32(mmax[d]));
  atomicAdd(&denom[d], w);
}

// agg[dst] += h[src] * alpha   (wave per edge, float4 per lane)
__global__ void k_edge_agg(const int* __restrict__ ei2, const float* __restrict__ als,
                           const float* __restrict__ ald,
                           const unsigned* __restrict__ mmax,
                           const float* __restrict__ denom,
                           const float* __restrict__ H, float* __restrict__ agg) {
  int gid  = blockIdx.x * 256 + threadIdx.x;
  int e    = gid >> 5;
  int lane = gid & 31;
  if (e >= ETOT) return;
  int s, d; edge_sd(ei2, e, s, d);
  float v = leaky(als[s] + ald[d]);
  float alpha = __expf(v - dec_f32(mmax[d])) / denom[d];
  const float4 h = ((const float4*)(H + (size_t)s * NHID))[lane];
  float* out = agg + (size_t)d * NHID + lane * 4;
  atomicAdd(out + 0, h.x * alpha);
  atomicAdd(out + 1, h.y * alpha);
  atomicAdd(out + 2, h.z * alpha);
  atomicAdd(out + 3, h.w * alpha);
}

// ---------------------------------------------------------------------------
// out = (relu(in + bias)) [/ max(||.||2, eps) if do_norm]   (wave per node)
// ---------------------------------------------------------------------------
__global__ void k_postproc(const float* __restrict__ in, const float* __restrict__ bias,
                           float* __restrict__ out, int do_norm) {
  int gid  = blockIdx.x * 256 + threadIdx.x;
  int node = gid >> 5, lane = gid & 31;
  if (node >= NN) return;
  float4 v = ((const float4*)(in + (size_t)node * NHID))[lane];
  if (bias) {
    const float4 b = ((const float4*)bias)[lane];
    v.x += b.x; v.y += b.y; v.z += b.z; v.w += b.w;
  }
  v.x = fmaxf(v.x, 0.f); v.y = fmaxf(v.y, 0.f);
  v.z = fmaxf(v.z, 0.f); v.w = fmaxf(v.w, 0.f);
  if (do_norm) {
    float ss = v.x * v.x + v.y * v.y + v.z * v.z + v.w * v.w;
#pragma unroll
    for (int m = 16; m > 0; m >>= 1) ss += __shfl_xor(ss, m, 32);
    float sc = 1.f / fmaxf(sqrtf(ss), EPSN);
    v.x *= sc; v.y *= sc; v.z *= sc; v.w *= sc;
  }
  ((float4*)(out + (size_t)node * NHID))[lane] = v;
}

// ---------------------------------------------------------------------------
// Fused MLP: score[i] (+)= relu(relu(x@W1+b1)@W2+b2)@W3+b3
// Block = 256 threads (8 waves) covers 16 nodes exactly. X and H1 live in
// LDS in bf16 fragment order; H2 stays f32 for the final dot product.
// ---------------------------------------------------------------------------
__global__ void k_mlp(const float* __restrict__ X,
                      const __bf16* __restrict__ W1f, const float* __restrict__ b1,
                      const __bf16* __restrict__ W2f, const float* __restrict__ b2,
                      const float* __restrict__ W3, const float* __restrict__ b3,
                      float* __restrict__ score, int init) {
  __shared__ __attribute__((aligned(32))) __bf16 Af[4 * 32 * 16];   // 4 KB
  __shared__ __attribute__((aligned(32))) __bf16 H1f[8 * 32 * 16];  // 8 KB
  __shared__ float H2[16 * 256];                                    // 16 KB
  int m0 = blockIdx.x * 16;
  int tid = threadIdx.x;
  for (int i = tid; i < 4 * 32 * 16; i += 256) {
    int j = i & 15, ln = (i >> 4) & 31, kk = i >> 9;
    int m = ln & 15;
    int k = kk * 32 + ((j >> 3) << 4) + ((ln >> 4) << 3) + (j & 7);
    Af[i] = (__bf16)X[(size_t)(m0 + m) * NHID + k];
  }
  __syncthreads();
  int wave = tid >> 5, lane = tid & 31;
  int col16 = lane & 15;
  int rbase = (lane >> 4) << 3;

  // stage 1: H1 = relu(A @ W1 + b1)   (K=128, C=256 -> 16 col tiles)
  for (int ct = wave; ct < 16; ct += 8) {
    v8f acc = {};
#pragma unroll
    for (int kk = 0; kk < 4; ++kk) {
      v16bf a = frag_load(&Af[(kk * 32 + lane) * 16]);
      v16bf b = frag_load(&W1f[((size_t)(ct * 4 + kk) * 32 + lane) * 16]);
      acc = wmma_bf16(a, b, acc);
    }
    int col = ct * 16 + col16;
    float bias = b1[col];
    // scatter into bf16 A-fragment layout for stage 2 (k = col)
    int kk2 = col >> 5, rem = col & 31;
    int j2  = ((rem >> 4) << 3) | (rem & 7);
    int lhb = ((rem >> 3) & 1) << 4;
#pragma unroll
    for (int r = 0; r < 8; ++r)
      H1f[((kk2 * 32 + (lhb | (rbase + r))) * 16) + j2] =
          (__bf16)fmaxf(acc[r] + bias, 0.f);
  }
  __syncthreads();

  // stage 2: H2 = relu(H1 @ W2 + b2)  (K=256)
  for (int ct = wave; ct < 16; ct += 8) {
    v8f acc = {};
#pragma unroll
    for (int kk = 0; kk < 8; ++kk) {
      v16bf a = frag_load(&H1f[(kk * 32 + lane) * 16]);
      v16bf b = frag_load(&W2f[((size_t)(ct * 8 + kk) * 32 + lane) * 16]);
      acc = wmma_bf16(a, b, acc);
    }
    int col = ct * 16 + col16;
    float bias = b2[col];
#pragma unroll
    for (int r = 0; r < 8; ++r)
      H2[(rbase + r) * 256 + col] = fmaxf(acc[r] + bias, 0.f);
  }
  __syncthreads();

  // stage 3: score = H2 @ W3 + b3
  if (tid < 16) {
    float s = 0.f;
    for (int k = 0; k < 256; ++k) s += H2[tid * 256 + k] * W3[k];
    s += b3[0];
    int row = m0 + tid;
    score[row] = (init ? 0.f : score[row]) + s;
  }
}

// ---------------------------------------------------------------------------
// Host orchestration
// ---------------------------------------------------------------------------
static void gat_layer(hipStream_t stream, const float* x_in, float* out,
                      const __bf16* Wf, const float* asl, const float* adl,
                      const float* bl, const int* ei2, float* h, float* agg,
                      float* als, float* ald, unsigned* mmax, float* denom,
                      int do_norm) {
  const int GB_FILLX = (NN * NHID + 255) / 256;
  const int GB_FILLN = (NN + 255) / 256;
  const int GB_NODEW = (NN * 32 + 255) / 256;
  const int GB_EDGE  = (ETOT + 255) / 256;
  const int GB_EDGEW = (ETOT * 32 + 255) / 256;
  const int GB_TILE  = NN / 16;

  k_fill_f32<<<GB_FILLX, 256, 0, stream>>>(agg, 0.f, NN * NHID);
  k_fill_f32<<<GB_FILLN, 256, 0, stream>>>(denom, 0.f, NN);
  k_fill_u32<<<GB_FILLN, 256, 0, stream>>>(mmax, 0u, NN);

  k_gemm128<<<GB_TILE, 256, 0, stream>>>(x_in, Wf, h);
  k_att_logits<<<GB_NODEW, 256, 0, stream>>>(h, asl, adl, als, ald);
  k_edge_max<<<GB_EDGE, 256, 0, stream>>>(ei2, als, ald, mmax);
  k_edge_sum<<<GB_EDGE, 256, 0, stream>>>(ei2, als, ald, mmax, denom);
  k_edge_agg<<<GB_EDGEW, 256, 0, stream>>>(ei2, als, ald, mmax, denom, h, agg);
  k_postproc<<<GB_NODEW, 256, 0, stream>>>(agg, bl, out, do_norm);
}

extern "C" void kernel_launch(void* const* d_in, const int* in_sizes, int n_in,
                              void* d_out, int out_size, void* d_ws, size_t ws_size,
                              hipStream_t stream) {
  const int*   ei1   = (const int*)d_in[0];
  const float* vals1 = (const float*)d_in[1];
  const int*   ei2   = (const int*)d_in[2];
  const float* winit = (const float*)d_in[3];
  const float* Wg    = (const float*)d_in[4];   // [7][128][128]
  const float* att_s = (const float*)d_in[5];   // [7][128]
  const float* att_d = (const float*)d_in[6];   // [7][128]
  const float* bg    = (const float*)d_in[7];   // [7][128]
  const float* W1    = (const float*)d_in[8];   // [128][256]
  const float* b1    = (const float*)d_in[9];
  const float* W2    = (const float*)d_in[10];  // [256][256]
  const float* b2    = (const float*)d_in[11];
  const float* W3    = (const float*)d_in[12];  // [256][1]
  const float* b3    = (const float*)d_in[13];
  float* score = (float*)d_out;

  // workspace: 3 x N*128 f32 feature buffers + 4 x N scalars + bf16 weights
  float* xA = (float*)d_ws;             // forward x / final x
  float* xB = xA + (size_t)NN * NHID;   // h = x @ W
  float* xC = xB + (size_t)NN * NHID;   // aggregation / xi
  float* als = xC + (size_t)NN * NHID;
  float* ald = als + NN;
  unsigned* mmax = (unsigned*)(ald + NN);
  float* denom = (float*)(mmax + NN);
  __bf16* Wgf = (__bf16*)(denom + NN);            // 7 * 16384 bf16
  __bf16* W1f = Wgf + (size_t)NGAT * NHID * NHID; // 32768 bf16
  __bf16* W2f = W1f + (size_t)NHID * 256;         // 65536 bf16

  const int GB_FILLX = (NN * NHID + 255) / 256;
  const int GB_NODEW = (NN * 32 + 255) / 256;
  const int GB_SPMM  = (NE1 * 32 + 255) / 256;
  const int GB_TILE  = NN / 16;

  // one-time weight packing into bf16 WMMA B-fragment layout
  for (int i = 0; i < NGAT; ++i)
    k_pack_w<<<(NHID * NHID + 255) / 256, 256, 0, stream>>>(
        Wg + (size_t)i * NHID * NHID, Wgf + (size_t)i * NHID * NHID, NHID, NHID);
  k_pack_w<<<(NHID * 256 + 255) / 256, 256, 0, stream>>>(W1, W1f, NHID, 256);
  k_pack_w<<<(256 * 256 + 255) / 256, 256, 0, stream>>>(W2, W2f, 256, 256);

  // init layer: x = l2norm(relu(spmm(adj1, weight_init)))
  k_fill_f32<<<GB_FILLX, 256, 0, stream>>>(xA, 0.f, NN * NHID);
  k_spmm<<<GB_SPMM, 256, 0, stream>>>(ei1, vals1, winit, xA);
  k_postproc<<<GB_NODEW, 256, 0, stream>>>(xA, nullptr, xA, 1);

  // 6 intermediate GAT layers (in place on xA)
  for (int i = 0; i < NINTER; ++i)
    gat_layer(stream, xA, xA, Wgf + (size_t)i * NHID * NHID, att_s + i * NHID,
              att_d + i * NHID, bg + i * NHID, ei2, xB, xC, als, ald, mmax,
              denom, 1);

  // score = mlp(x_final)
  k_mlp<<<GB_TILE, 256, 0, stream>>>(xA, W1f, b1, W2f, b2, W3, b3, score, 1);

  // score += mlp(l2norm(relu(gat_i(x_final))))  for i = 0..5
  for (int i = 0; i < NINTER; ++i) {
    gat_layer(stream, xA, xC, Wgf + (size_t)i * NHID * NHID, att_s + i * NHID,
              att_d + i * NHID, bg + i * NHID, ei2, xB, xC, als, ald, mmax,
              denom, 1);
    k_mlp<<<GB_TILE, 256, 0, stream>>>(xC, W1f, b1, W2f, b2, W3, b3, score, 0);
  }

  // score += mlp(relu(gat_last(x_final)))   (no l2norm on last layer)
  gat_layer(stream, xA, xC, Wgf + (size_t)NINTER * NHID * NHID,
            att_s + NINTER * NHID, att_d + NINTER * NHID, bg + NINTER * NHID,
            ei2, xB, xC, als, ald, mmax, denom, 0);
  k_mlp<<<GB_TILE, 256, 0, stream>>>(xC, W1f, b1, W2f, b2, W3, b3, score, 0);
}
